// GDKVM_14705968021664
// MI455X (gfx1250) — compile-verified, matrix operand order checked by reading, not run on hardware
//
#include <hip/hip_runtime.h>

typedef unsigned short u16;
typedef __attribute__((ext_vector_type(4)))  unsigned int u32x4;
typedef __attribute__((ext_vector_type(16))) __bf16      v16bf;
typedef __attribute__((ext_vector_type(8)))  float       v8f;

#define USE_ASYNC_LDS 1

union ABfrag { u32x4 u[2]; v16bf v; };

__device__ __forceinline__ u16 f2bf(float f) {
    unsigned int u = __float_as_uint(f);
    unsigned int r = (u + 0x7FFFu + ((u >> 16) & 1u)) >> 16;   // round-to-nearest-even
    return (u16)r;
}

__device__ __forceinline__ unsigned lds_addr32(const void* p) {
    // Generic LDS pointer: low 32 bits are the LDS byte offset (ISA aperture rule).
    return (unsigned)(unsigned long long)p;
}

// ---------------------------------------------------------------------------
// bf16 WMMA GEMM:  C[M,N] = alpha * A[M,K] x B[K,N] (+ bias[m])
// A   : row-major [Mp,K] bf16, Mp = ceil(M/128)*128 rows allocated (pad rows zero)
// Bt  : row-major [Np,K] bf16 == B^T, Np = gridDim.x*128 rows (pad rows zero)
// C   : row-major [M,N]  fp32 (stores guarded by m<M, n<N)
// Block: 256 threads = 8 waves; C tile 128x128; each wave 64x32 (4x2 WMMAs).
// K staged through LDS in 32-wide slices via GLOBAL_LOAD_ASYNC_TO_LDS_B128.
// ---------------------------------------------------------------------------
__global__ __launch_bounds__(256) void k_gemm_bf16(
    const u16* __restrict__ A, const u16* __restrict__ Bt,
    float* __restrict__ C, const float* __restrict__ bias,
    int M, int N, int K, float alpha)
{
    __shared__ u32x4 As4[128 * 5];   // 128 rows x 32 bf16, stride 5 chunks (pad)
    __shared__ u32x4 Bs4[128 * 5];

    const int tid  = threadIdx.x;
    const int lane = tid & 31;
    const int wave = tid >> 5;
    const int wm = (wave >> 2) * 64;      // wave row offset in 128x128 tile {0,64}
    const int wn = (wave & 3) * 32;       // wave col offset {0,32,64,96}

    const long rowBase = (long)blockIdx.y * 128;   // A rows / C rows
    const long colBase = (long)blockIdx.x * 128;   // Bt rows / C cols

    const int r0 = tid >> 1;   // 0..127 : tile row this thread stages
    const int c0 = tid & 1;    // chunks c0 and c0+2 (16B each)

    const u16* Arow = A  + (rowBase + r0) * (long)K + 8 * c0;
    const u16* Brow = Bt + (colBase + r0) * (long)K + 8 * c0;

#if USE_ASYNC_LDS
    const unsigned ldsA = lds_addr32(&As4[r0 * 5 + c0]);
    const unsigned ldsB = lds_addr32(&Bs4[r0 * 5 + c0]);
#endif

    v8f acc[4][2] = {};

    const int h  = lane >> 4;
    const int rr = lane & 15;

    for (int kt = 0; kt < K; kt += 32) {
#if USE_ASYNC_LDS
        {
            unsigned long long ga = (unsigned long long)(Arow + kt);
            unsigned long long gb = (unsigned long long)(Brow + kt);
            asm volatile("global_load_async_to_lds_b128 %0, %1, off"
                         :: "v"(ldsA), "v"(ga) : "memory");
            asm volatile("global_load_async_to_lds_b128 %0, %1, off offset:32"
                         :: "v"(ldsA), "v"(ga) : "memory");
            asm volatile("global_load_async_to_lds_b128 %0, %1, off"
                         :: "v"(ldsB), "v"(gb) : "memory");
            asm volatile("global_load_async_to_lds_b128 %0, %1, off offset:32"
                         :: "v"(ldsB), "v"(gb) : "memory");
        }
#else
        {
            const u32x4* ag = (const u32x4*)(Arow + kt);
            const u32x4* bg = (const u32x4*)(Brow + kt);
            As4[r0 * 5 + c0]     = ag[0];
            As4[r0 * 5 + c0 + 2] = ag[2];
            Bs4[r0 * 5 + c0]     = bg[0];
            Bs4[r0 * 5 + c0 + 2] = bg[2];
        }
#endif
        if (kt + 32 < K) {              // uniform; prefetch next K slice
            __builtin_prefetch(Arow + kt + 32, 0, 1);
            __builtin_prefetch(Brow + kt + 32, 0, 1);
        }
#if USE_ASYNC_LDS
        asm volatile("s_wait_asynccnt 0x0" ::: "memory");
#endif
        __syncthreads();

        ABfrag af[4], bf[2];
#pragma unroll
        for (int i = 0; i < 4; ++i) {
            int r = wm + i * 16 + rr;
            af[i].u[0] = As4[r * 5 + h];
            af[i].u[1] = As4[r * 5 + 2 + h];
        }
#pragma unroll
        for (int j = 0; j < 2; ++j) {
            int r = wn + j * 16 + rr;
            bf[j].u[0] = Bs4[r * 5 + h];
            bf[j].u[1] = Bs4[r * 5 + 2 + h];
        }
#pragma unroll
        for (int i = 0; i < 4; ++i)
#pragma unroll
            for (int j = 0; j < 2; ++j)
                acc[i][j] = __builtin_amdgcn_wmma_f32_16x16x32_bf16(
                    false, af[i].v, false, bf[j].v,
                    (short)0, acc[i][j], false, false);
        __syncthreads();
    }

    // C layout per ISA: VGPR r, lane l -> m = r + 8*(l>>4), n = l&15
    const int nn = lane & 15;
#pragma unroll
    for (int i = 0; i < 4; ++i) {
#pragma unroll
        for (int j = 0; j < 2; ++j) {
            long n = colBase + wn + j * 16 + nn;
            if (n < N) {
                long m0 = rowBase + wm + i * 16 + 8 * h;
#pragma unroll
                for (int r = 0; r < 8; ++r) {
                    long m = m0 + r;
                    if (m < M) {
                        float v = alpha * acc[i][j][r];
                        if (bias) v += bias[m];
                        C[m * (long)N + n] = v;
                    }
                }
            }
        }
    }
}

// ---------------------------------------------------------------------------
__global__ void k_f32_to_bf16(const float* __restrict__ in, u16* __restrict__ out, long n)
{
    long i = (long)blockIdx.x * 256 + threadIdx.x;
    if (i < n) out[i] = f2bf(in[i]);
}

__global__ void k_zero_u16(u16* __restrict__ out, long n)
{
    long i = (long)blockIdx.x * 256 + threadIdx.x;
    if (i < n) out[i] = 0;
}

// in [C, P] fp32 -> out [Np, C] bf16 (rows p >= P zero-filled)
__global__ void k_transpose_pad(const float* __restrict__ in, u16* __restrict__ out,
                                int C, int P, long total)
{
    long i = (long)blockIdx.x * 256 + threadIdx.x;
    if (i >= total) return;
    int c = (int)(i % C);
    int p = (int)(i / C);
    out[i] = (p < P) ? f2bf(in[(long)c * P + p]) : (u16)0;
}

// 3x3 SAME im2col, transposed + bf16: col[p, c*9 + ky*3 + kx]
__global__ void k_im2col3x3(const float* __restrict__ src, const float* __restrict__ src2,
                            int C1, int Ctot, u16* __restrict__ col, long total)
{
    long i = (long)blockIdx.x * 256 + threadIdx.x;
    if (i >= total) return;
    int K9  = Ctot * 9;
    int p   = (int)(i / K9);
    int rem = (int)(i % K9);
    int c = rem / 9, k = rem % 9;
    u16 o = 0;
    if (p < 784) {
        int y = p / 28, x = p % 28;
        int yy = y + k / 3 - 1, xx = x + k % 3 - 1;
        if (yy >= 0 && yy < 28 && xx >= 0 && xx < 28) {
            float v = (c < C1) ? src [(long)c        * 784 + yy * 28 + xx]
                               : src2[(long)(c - C1) * 784 + yy * 28 + xx];
            o = f2bf(v);
        }
    }
    col[i] = o;
}

// Column softmax of L[M, P] over m; writes affT[Np, M] bf16
__global__ void k_softmax_col(const float* __restrict__ L, u16* __restrict__ affT,
                              int M, int P)
{
    const int p = blockIdx.x;
    const int tid = threadIdx.x;
    __shared__ float red[256];
    u16* orow = affT + (long)p * M;
    if (p >= P) {
        for (int m = tid; m < M; m += 256) orow[m] = 0;
        return;
    }
    const float* col = L + p;
    float mx = -3.4e38f;
    for (int m = tid; m < M; m += 256) mx = fmaxf(mx, col[(long)m * P]);
    red[tid] = mx; __syncthreads();
    for (int s = 128; s > 0; s >>= 1) { if (tid < s) red[tid] = fmaxf(red[tid], red[tid + s]); __syncthreads(); }
    mx = red[0]; __syncthreads();
    float sum = 0.f;
    for (int m = tid; m < M; m += 256) sum += __expf(col[(long)m * P] - mx);
    red[tid] = sum; __syncthreads();
    for (int s = 128; s > 0; s >>= 1) { if (tid < s) red[tid] += red[tid + s]; __syncthreads(); }
    float inv = 1.0f / red[0];
    for (int m = tid; m < M; m += 256)
        orow[m] = f2bf(__expf(col[(long)m * P] - mx) * inv);
}

// Gated recurrent update
__global__ void k_gate(const float* __restrict__ vals, const float* __restrict__ h,
                       float* __restrict__ out, long total, int CsP)
{
    long i = (long)blockIdx.x * 256 + threadIdx.x;
    if (i >= total) return;
    long bn  = i / CsP;
    long rem = i % CsP;
    const float* v = vals + bn * (long)(3 * CsP);
    float f  = 1.0f / (1.0f + __expf(-v[rem]));
    float u  = 1.0f / (1.0f + __expf(-v[rem + CsP]));
    float nv = tanhf(v[rem + 2 * CsP]);
    out[i] = f * h[i] * (1.0f - u) + u * nv;
}

// ---------------------------------------------------------------------------
extern "C" void kernel_launch(void* const* d_in, const int* in_sizes, int n_in,
                              void* d_out, int out_size, void* d_ws, size_t ws_size,
                              hipStream_t stream) {
    (void)in_sizes; (void)n_in; (void)out_size; (void)ws_size;
    constexpr int B = 2, N = 3, P = 784, M = 12544;
    constexpr int Cp = 1024, Ch = 512, Ck = 64, Cv = 512, Cs = 256;
    constexpr int Np = 896;                 // P padded to multiple of 128 (7 tiles)
    constexpr int K2 = Ch * 9;              // 4608
    constexpr int K3 = (Cv + Cs) * 9;       // 6912
    constexpr int Co3 = 3 * Cs;             // 768

    const float* f16 = (const float*)d_in[0];
    const float* mk  = (const float*)d_in[1];
    const float* mv  = (const float*)d_in[2];
    const float* h   = (const float*)d_in[3];
    const float* w1  = (const float*)d_in[4];
    const float* b1  = (const float*)d_in[5];
    const float* w2  = (const float*)d_in[6];
    const float* b2  = (const float*)d_in[7];
    const float* wt  = (const float*)d_in[8];
    const float* bt  = (const float*)d_in[9];
    float* out = (float*)d_out;

    char* ws = (char*)d_ws;
    size_t off = 0;
    auto alloc = [&](size_t bytes) -> void* {
        void* p = ws + off;
        off = (off + bytes + 255) & ~(size_t)255;
        return p;
    };
    u16*   mv_bf   = (u16*)  alloc((size_t)B * N * Cv * M * 2);
    u16*   w1_bf   = (u16*)  alloc((size_t)Ch * Cp * 2);
    u16*   w2_bf   = (u16*)  alloc((size_t)128 * K2 * 2);         // Ck=64 padded to 128 rows
    u16*   wt_bf   = (u16*)  alloc((size_t)Co3 * K3 * 2);
    u16*   XT      = (u16*)  alloc((size_t)B * Np * Cp * 2);
    u16*   mkT     = (u16*)  alloc((size_t)B * M * Ck * 2);
    float* Y1      = (float*)alloc((size_t)B * Ch * P * 4);
    u16*   col2    = (u16*)  alloc((size_t)Np * K2 * 2);          // reused per batch
    float* Y2      = (float*)alloc((size_t)B * Ck * P * 4);
    u16*   qkT     = (u16*)  alloc((size_t)B * Np * Ck * 2);
    float* logits  = (float*)alloc((size_t)M * P * 4);            // reused per batch
    u16*   affT    = (u16*)  alloc((size_t)B * Np * M * 2);
    float* rdout   = (float*)alloc((size_t)B * N * Cv * P * 4);
    u16*   col3    = (u16*)  alloc((size_t)Np * K3 * 2);          // reused per (b,n)
    float* vals    = (float*)alloc((size_t)B * N * Co3 * P * 4);

    auto blks = [](long n) { return (unsigned)((n + 255) / 256); };
    auto gemm = [&](const u16* A, const u16* Bt, float* C, const float* bias,
                    int Mm, int Kk, float alpha) {
        dim3 g(Np / 128, (unsigned)((Mm + 127) / 128));
        k_gemm_bf16<<<g, 256, 0, stream>>>(A, Bt, C, bias, Mm, P, Kk, alpha);
    };

    // --- weight / memory conversions to bf16 ---
    k_f32_to_bf16<<<blks((long)Ch * Cp),        256, 0, stream>>>(w1, w1_bf, (long)Ch * Cp);
    k_f32_to_bf16<<<blks((long)Ck * K2),        256, 0, stream>>>(w2, w2_bf, (long)Ck * K2);
    k_zero_u16   <<<blks((long)(128 - Ck) * K2),256, 0, stream>>>(w2_bf + (size_t)Ck * K2,
                                                                  (long)(128 - Ck) * K2);
    k_f32_to_bf16<<<blks((long)Co3 * K3),       256, 0, stream>>>(wt, wt_bf, (long)Co3 * K3);
    k_f32_to_bf16<<<blks((long)B * N * Cv * M), 256, 0, stream>>>(mv, mv_bf, (long)B * N * Cv * M);
    for (int b = 0; b < B; ++b) {
        k_transpose_pad<<<blks((long)Np * Cp), 256, 0, stream>>>(
            f16 + (size_t)b * Cp * P, XT + (size_t)b * Np * Cp, Cp, P, (long)Np * Cp);
        k_transpose_pad<<<blks((long)M * Ck), 256, 0, stream>>>(
            mk + (size_t)b * Ck * M, mkT + (size_t)b * M * Ck, Ck, M, (long)M * Ck);
    }

    // --- Stage 1: 1x1 conv ---
    for (int b = 0; b < B; ++b)
        gemm(w1_bf, XT + (size_t)b * Np * Cp, Y1 + (size_t)b * Ch * P, b1, Ch, Cp, 1.0f);

    // --- Stage 2: 3x3 conv (im2col) + transpose for logits GEMM ---
    for (int b = 0; b < B; ++b) {
        k_im2col3x3<<<blks((long)Np * K2), 256, 0, stream>>>(
            Y1 + (size_t)b * Ch * P, Y1 + (size_t)b * Ch * P, Ch, Ch, col2, (long)Np * K2);
        gemm(w2_bf, col2, Y2 + (size_t)b * Ck * P, b2, Ck, K2, 1.0f);
        k_transpose_pad<<<blks((long)Np * Ck), 256, 0, stream>>>(
            Y2 + (size_t)b * Ck * P, qkT + (size_t)b * Np * Ck, Ck, P, (long)Np * Ck);
    }

    // --- Stage 3: logits + softmax over memory axis ---
    for (int b = 0; b < B; ++b) {
        gemm(mkT + (size_t)b * M * Ck, qkT + (size_t)b * Np * Ck, logits, nullptr,
             M, Ck, 0.125f);
        k_softmax_col<<<Np, 256, 0, stream>>>(logits, affT + (size_t)b * Np * M, M, P);
    }

    // --- Stage 4: readout (the 60-GFLOP GEMM) ---
    for (int b = 0; b < B; ++b)
        for (int n = 0; n < N; ++n) {
            size_t bn = (size_t)b * N + n;
            gemm(mv_bf + bn * (size_t)Cv * M, affT + (size_t)b * Np * M,
                 rdout + bn * (size_t)Cv * P, nullptr, Cv, M, 1.0f);
        }

    // --- Stage 5: gate conv ---
    for (int b = 0; b < B; ++b)
        for (int n = 0; n < N; ++n) {
            size_t bn = (size_t)b * N + n;
            k_im2col3x3<<<blks((long)Np * K3), 256, 0, stream>>>(
                rdout + bn * (size_t)Cv * P, h + bn * (size_t)Cs * P,
                Cv, Cv + Cs, col3, (long)Np * K3);
            gemm(wt_bf, col3, vals + bn * (size_t)Co3 * P, bt, Co3, K3, 1.0f);
        }

    // --- Stage 6: gated recurrent update ---
    long total = (long)B * N * Cs * P;
    k_gate<<<blks(total), 256, 0, stream>>>(vals, h, out, total, Cs * P);
}
